// GCLSTM_21784074125834
// MI455X (gfx1250) — matrix operand order repaired and checked by reading, not compiled, not run on hardware
//
#include <hip/hip_runtime.h>
#include <hip/hip_bf16.h>

#define NNODES 50000
#define NEDGES 1600000
#define FDIM   128
#define GR     128     // rows per workgroup in the WMMA kernel (8 waves x 16)

#if defined(__has_builtin)
#if __has_builtin(__builtin_amdgcn_global_load_async_to_lds_b128) && \
    __has_builtin(__builtin_amdgcn_s_wait_asynccnt)
#define USE_ASYNC_LDS 1
#endif
#endif
#ifndef USE_ASYNC_LDS
#define USE_ASYNC_LDS 0
#endif

typedef __attribute__((ext_vector_type(16))) __bf16 v16bf;
typedef __attribute__((ext_vector_type(8)))  float  v8f;
typedef __attribute__((ext_vector_type(4)))  int    v4i;

union BF16x16 { v16bf v; uint4 q[2]; unsigned short s[16]; };

__device__ __forceinline__ unsigned short f2bf(float f) {
    unsigned int u = __float_as_uint(f);
    unsigned int r = u + 0x7FFFu + ((u >> 16) & 1u);   // round-to-nearest-even
    return (unsigned short)(r >> 16);
}
__device__ __forceinline__ float sigm(float x) { return 1.0f / (1.0f + __expf(-x)); }
__device__ __forceinline__ float ftanh(float x) {
    float e = __expf(2.0f * x);
    return (e - 1.0f) / (e + 1.0f);
}

// ---------------- helper kernels ----------------

__global__ void zero_f32(float* __restrict__ p, long n) {
    long i = (long)blockIdx.x * blockDim.x + threadIdx.x;
    if (i < n) p[i] = 0.0f;
}

// Wx (4,3,128,128) f32 row-major [k][n]  ->  Wt bf16 [mat][n][k]
__global__ void wprep_k(const float* __restrict__ Wx, unsigned short* __restrict__ Wt) {
    int i = blockIdx.x * blockDim.x + threadIdx.x;
    if (i >= 4 * 3 * FDIM * FDIM) return;
    int mat = i >> 14;            // 128*128 = 16384 per matrix
    int j   = i & 16383;
    int k   = j >> 7;
    int n   = j & 127;
    Wt[mat * 16384 + n * FDIM + k] = f2bf(Wx[i]);
}

__global__ void deg_k(const int* __restrict__ row, const float* __restrict__ w,
                      float* __restrict__ deg) {
    int e = blockIdx.x * blockDim.x + threadIdx.x;
    if (e < NEDGES) atomicAdd(deg + row[e], w[e]);
}

__global__ void dis_k(float* __restrict__ deg) {
    int i = blockIdx.x * blockDim.x + threadIdx.x;
    if (i < NNODES) {
        float d = deg[i];
        deg[i] = (d > 0.0f) ? rsqrtf(fmaxf(d, 1e-30f)) : 0.0f;
    }
}

__global__ void norm_k(const int* __restrict__ row, const int* __restrict__ col,
                       const float* __restrict__ w, const float* __restrict__ dis,
                       float* __restrict__ nrm) {
    int e = blockIdx.x * blockDim.x + threadIdx.x;
    if (e < NEDGES) nrm[e] = -dis[row[e]] * w[e] * dis[col[e]];
}

// dst[row[e]] += nrm[e] * src[col[e]]   (one 32-lane group per edge, 4 floats/lane)
__global__ void prop_k(const float* __restrict__ src, const float* __restrict__ nrm,
                       const int* __restrict__ row, const int* __restrict__ col,
                       float* __restrict__ dst) {
    long idx = (long)blockIdx.x * blockDim.x + threadIdx.x;
    int e = (int)(idx >> 5);
    int l = (int)(idx & 31) << 2;
    if (e >= NEDGES) return;
    float nv = nrm[e];
    const float4 v = *(const float4*)(src + (long)col[e] * FDIM + l);
    float* d = dst + (long)row[e] * FDIM + l;
    atomicAdd(d + 0, nv * v.x);
    atomicAdd(d + 1, nv * v.y);
    atomicAdd(d + 2, nv * v.z);
    atomicAdd(d + 3, nv * v.w);
}

__global__ void tx2_k(float* __restrict__ p2, const float* __restrict__ x, long n) {
    long i = (long)blockIdx.x * blockDim.x + threadIdx.x;
    if (i < n) p2[i] = 2.0f * p2[i] - x[i];
}

// ---------------- fused WMMA gate kernel ----------------

// Copy one gate's (pre-transposed bf16) weights global -> LDS.
// Async path: issue GLOBAL_LOAD_ASYNC_TO_LDS_B128 and return immediately
// (completion enforced later via s_wait_asynccnt + barrier).
__device__ __forceinline__ void stageW(const unsigned short* __restrict__ Wt, int g,
                                       unsigned short* dst, int t) {
    const unsigned short* src = Wt + g * 3 * FDIM * FDIM;
#if USE_ASYNC_LDS
    for (int i = t; i < 3 * FDIM * FDIM / 8; i += 256) {
        void* sp = (void*)(src + 8 * i);   // strip const in generic AS first
        void* dp = (void*)(dst + 8 * i);
        __builtin_amdgcn_global_load_async_to_lds_b128(
            (__attribute__((address_space(1))) v4i*)sp,
            (__attribute__((address_space(3))) v4i*)dp,
            0, 0);
    }
#else
    const uint4* s4 = (const uint4*)src;
    uint4* d4 = (uint4*)dst;
    for (int i = t; i < 3 * FDIM * FDIM / 8; i += 256) d4[i] = s4[i];
#endif
}

__device__ __forceinline__ void waitAsync() {
#if USE_ASYNC_LDS
    __builtin_amdgcn_s_wait_asynccnt(0);
#endif
}

__device__ __forceinline__ void loadB(BF16x16 (&b)[8], const unsigned short* __restrict__ sW,
                                      int idx, int colBase, int half) {
    int km = idx >> 2, kk = idx & 3;
    const int K0 = kk * 32;
    const unsigned short* wk = sW + km * (FDIM * FDIM) + colBase + K0 + half * 8;
#pragma unroll
    for (int ct = 0; ct < 8; ++ct) {
        const unsigned short* bp = wk + ct * 16 * FDIM;
        b[ct].q[0] = *(const uint4*)(bp);
        b[ct].q[1] = *(const uint4*)(bp + 16);
    }
}

// acc += sum_{km} A_lds[km] (16x128, bf16) @ W_lds[km] (128x128, bf16, stored [n][k])
// B fragments are double-buffered so 8 WMMAs cover each DS-load batch.
__device__ __forceinline__ void gate_mm(v8f (&acc)[8],
                                        const unsigned short* __restrict__ sA,
                                        int rowOff,
                                        const unsigned short* __restrict__ sW,
                                        int colBase, int half) {
    BF16x16 b[2][8];
    loadB(b[0], sW, 0, colBase, half);
#pragma unroll
    for (int idx = 0; idx < 12; ++idx) {
        if (idx + 1 < 12) loadB(b[(idx + 1) & 1], sW, idx + 1, colBase, half);
        const int km = idx >> 2, kk = idx & 3, K0 = kk * 32;
        const unsigned short* sAk = sA + km * (GR * FDIM) + rowOff;
        BF16x16 a;
        a.q[0] = *(const uint4*)(sAk + K0 + half * 8);        // K0 + half*8 .. +7
        a.q[1] = *(const uint4*)(sAk + K0 + 16 + half * 8);   // K0+16+half*8 .. +7
#pragma unroll
        for (int ct = 0; ct < 8; ++ct) {
            acc[ct] = __builtin_amdgcn_wmma_f32_16x16x32_bf16(
                false, a.v, false, b[idx & 1][ct].v, (short)0, acc[ct], false, false);
        }
    }
}

__global__ __launch_bounds__(256, 1)
void gates_wmma(const float* __restrict__ X, const float* __restrict__ T1,
                const float* __restrict__ T2, const unsigned short* __restrict__ Wt,
                const float* __restrict__ bx, const float* __restrict__ bh,
                const float* __restrict__ bg, const float* __restrict__ wc,
                float* __restrict__ out) {
    extern __shared__ __align__(16) unsigned short smem[];
    unsigned short* A  = smem;                        // 3*GR*FDIM bf16   = 96 KB
    unsigned short* W0 = smem + 3 * GR * FDIM;        // gate weights     = 96 KB
    unsigned short* W1 = W0 + 3 * FDIM * FDIM;        // ping-pong buffer = 96 KB

    const int t     = threadIdx.x;
    const int wgRow = blockIdx.x * GR;

    // Kick off async staging of gate-0 weights, then convert A tiles while it flies.
    stageW(Wt, 0, W0, t);

    // Stage Tx0/Tx1/Tx2 row block into LDS as bf16 (row-major)
    for (int i = t; i < GR * 32; i += 256) {        // one float4 (4 cols) per iter
        int r  = i >> 5;
        int c4 = (i & 31) << 2;
        int grow = wgRow + r;
        float4 v0 = make_float4(0, 0, 0, 0), v1 = v0, v2 = v0;
        if (grow < NNODES) {
            long o = (long)grow * FDIM + c4;
            v0 = *(const float4*)(X  + o);
            v1 = *(const float4*)(T1 + o);
            v2 = *(const float4*)(T2 + o);
        }
        unsigned short* d0 = A + r * FDIM + c4;
        uint2 p;
        p.x = (unsigned)f2bf(v0.x) | ((unsigned)f2bf(v0.y) << 16);
        p.y = (unsigned)f2bf(v0.z) | ((unsigned)f2bf(v0.w) << 16);
        *(uint2*)(d0) = p;
        p.x = (unsigned)f2bf(v1.x) | ((unsigned)f2bf(v1.y) << 16);
        p.y = (unsigned)f2bf(v1.z) | ((unsigned)f2bf(v1.w) << 16);
        *(uint2*)(d0 + GR * FDIM) = p;
        p.x = (unsigned)f2bf(v2.x) | ((unsigned)f2bf(v2.y) << 16);
        p.y = (unsigned)f2bf(v2.z) | ((unsigned)f2bf(v2.w) << 16);
        *(uint2*)(d0 + 2 * GR * FDIM) = p;
    }
    waitAsync();
    __syncthreads();

    const int wave = t >> 5, lane = t & 31;
    const int m = lane & 15, half = lane >> 4;
    const int rowOff  = (wave * 16 + m) * FDIM;
    const int colBase = m * FDIM;

    v8f z;
#pragma unroll
    for (int i = 0; i < 8; ++i) z[i] = 0.0f;

    v8f acc[8], C[8];

    // ---- input gate I (weights in W0); gate-2 weights stream into W1 meanwhile ----
    stageW(Wt, 2, W1, t);
#pragma unroll
    for (int ct = 0; ct < 8; ++ct) acc[ct] = z;
    gate_mm(acc, A, rowOff, W0, colBase, half);
#pragma unroll
    for (int ct = 0; ct < 8; ++ct) {
        int n = ct * 16 + m;
        float bs = bx[0 * FDIM + n] + bh[0 * FDIM + n] + bg[0 * FDIM + n];
#pragma unroll
        for (int r = 0; r < 8; ++r) C[ct][r] = sigm(acc[ct][r] + bs);   // I (temp)
    }
    waitAsync();
    __syncthreads();   // all waves done reading W0; W1 fully populated

    // ---- candidate (weights in W1); gate-3 weights stream into W0 meanwhile ----
    stageW(Wt, 3, W0, t);
#pragma unroll
    for (int ct = 0; ct < 8; ++ct) acc[ct] = z;
    gate_mm(acc, A, rowOff, W1, colBase, half);
#pragma unroll
    for (int ct = 0; ct < 8; ++ct) {
        int n = ct * 16 + m;
        float bs = bx[2 * FDIM + n] + bh[2 * FDIM + n] + bg[2 * FDIM + n];
#pragma unroll
        for (int r = 0; r < 8; ++r) C[ct][r] = C[ct][r] * ftanh(acc[ct][r] + bs);
    }
    waitAsync();
    __syncthreads();

    // ---- output gate (weights in W0): H = O*tanh(C), relu, store ----
#pragma unroll
    for (int ct = 0; ct < 8; ++ct) acc[ct] = z;
    gate_mm(acc, A, rowOff, W0, colBase, half);
#pragma unroll
    for (int ct = 0; ct < 8; ++ct) {
        int n = ct * 16 + m;
        float bs  = bx[3 * FDIM + n] + bh[3 * FDIM + n] + bg[3 * FDIM + n];
        float wc2 = wc[2 * FDIM + n];
#pragma unroll
        for (int r = 0; r < 8; ++r) {
            float cv = C[ct][r];
            float o  = sigm(acc[ct][r] + bs + wc2 * cv);
            float h  = fmaxf(o * ftanh(cv), 0.0f);
            int grow = wgRow + wave * 16 + half * 8 + r;
            if (grow < NNODES) out[(long)grow * FDIM + n] = h;
        }
    }
}

// ---------------- launcher ----------------

extern "C" void kernel_launch(void* const* d_in, const int* in_sizes, int n_in,
                              void* d_out, int out_size, void* d_ws, size_t ws_size,
                              hipStream_t stream) {
    (void)in_sizes; (void)n_in; (void)out_size; (void)ws_size;
    const long N = NNODES, E = NEDGES;

    // Only timestep t = T-1 = 2 matters (scan carry is unused by the step).
    const float* X    = (const float*)d_in[0] + 2L * N * FDIM;   // node_feats[2]
    const float* w    = (const float*)d_in[1] + 2L * E;          // edge_feats[2]
    const float* Wx   = (const float*)d_in[2];                   // (4,3,128,128)
    const float* bx   = (const float*)d_in[3];
    const float* bh   = (const float*)d_in[5];
    const float* wc   = (const float*)d_in[6];
    const float* bg   = (const float*)d_in[7];
    const int*   ei   = (const int*)d_in[8] + 2L * 2 * E;        // edge_index[2]
    const int*   row  = ei;
    const int*   col  = ei + E;
    float*       out  = (float*)d_out;

    // workspace carve (all 16B-aligned by construction)
    float* deg = (float*)d_ws;                         // N
    float* nrm = deg + N;                              // E
    float* t1  = nrm + E;                              // N*128
    float* p2  = t1 + N * (long)FDIM;                  // N*128
    unsigned short* wt = (unsigned short*)(p2 + N * (long)FDIM);  // 4*3*128*128 bf16

    const int B = 256;
    zero_f32<<<dim3((unsigned)((N + B - 1) / B)), B, 0, stream>>>(deg, N);
    zero_f32<<<dim3((unsigned)((2 * N * FDIM + B - 1) / B)), B, 0, stream>>>(t1, 2 * N * FDIM);
    wprep_k<<<dim3((4 * 3 * FDIM * FDIM + B - 1) / B), B, 0, stream>>>(Wx, wt);

    deg_k<<<dim3((unsigned)((E + B - 1) / B)), B, 0, stream>>>(row, w, deg);
    dis_k<<<dim3((unsigned)((N + B - 1) / B)), B, 0, stream>>>(deg);
    norm_k<<<dim3((unsigned)((E + B - 1) / B)), B, 0, stream>>>(row, col, w, deg, nrm);

    const long propThreads = E * 32;
    prop_k<<<dim3((unsigned)((propThreads + B - 1) / B)), B, 0, stream>>>(X, nrm, row, col, t1);
    prop_k<<<dim3((unsigned)((propThreads + B - 1) / B)), B, 0, stream>>>(t1, nrm, row, col, p2);
    tx2_k<<<dim3((unsigned)((N * FDIM + B - 1) / B)), B, 0, stream>>>(p2, X, N * FDIM);

    // 288 KB dynamic LDS: A-tiles (96 KB) + ping-pong gate weights (2 x 96 KB).
    // CDNA5 allows up to 320 KB LDS per workgroup.
    const unsigned ldsBytes = (3 * GR * FDIM + 2 * 3 * FDIM * FDIM) * 2;
    gates_wmma<<<dim3((unsigned)((N + GR - 1) / GR)), 256, ldsBytes, stream>>>(
        X, t1, p2, wt, bx, bh, bg, wc, out);
}
